// WaveNetModel_11579231830326
// MI455X (gfx1250) — matrix operand order repaired and checked by reading, not compiled, not run on hardware
//
#include <hip/hip_runtime.h>

// ---------------------------------------------------------------------------
// WaveNet forward on gfx1250 (MI455X), wave32, v_wmma_f32_16x16x32_bf16 +
// tensor_load_to_lds (TDM) staging for the skip GEMM.
// Canonical activation layout: X[c][n], n = l*64 + b  (batch fastest).
// Dilated conv == 3-tap conv along n with offset +-d*64, clamped to [0,65536).
// WMMA B operands use a pair-interleaved bf16 layout [c/2][n][2] so that one
// aligned b32 load fills one fragment dword (rows k,k+1 at one column).
// ---------------------------------------------------------------------------

typedef __bf16 bf16;
typedef __attribute__((ext_vector_type(16))) __bf16        v16bf;
typedef __attribute__((ext_vector_type(8)))  float         v8f;
typedef __attribute__((ext_vector_type(8)))  unsigned int  v8u;
typedef __attribute__((ext_vector_type(4)))  unsigned int  v4u;
typedef __attribute__((ext_vector_type(8)))  int           v8i;
typedef __attribute__((ext_vector_type(4)))  int           v4i;

union FragBF {
  v16bf bf;
  v8u   u;
};

#define WMMA_BF16(a, b, c) \
  __builtin_amdgcn_wmma_f32_16x16x32_bf16(false, (a), false, (b), (short)0, (c), false, false)

#if defined(__has_builtin)
#if __has_builtin(__builtin_amdgcn_tensor_load_to_lds) && \
    __has_builtin(__builtin_amdgcn_s_wait_tensorcnt)
#define CDNA5_HAS_TDM 1
#endif
#endif
#ifndef CDNA5_HAS_TDM
#define CDNA5_HAS_TDM 0
#endif

static constexpr int BATCH  = 64;
static constexpr int LEN    = 1024;
static constexpr int NTOT   = BATCH * LEN;   // 65536 columns
static constexpr int RC     = 32;
static constexpr int DC     = 32;
static constexpr int SC     = 256;
static constexpr int NLAYER = 40;
static constexpr int CLS    = 256;
static constexpr int KFC    = SC * LEN;      // 262144

// Branch-free transcendentals on the gfx1250 TRANS pipe (co-executes w/ WMMA).
__device__ inline float fast_sigmoid(float x) {
#if defined(__has_builtin) && __has_builtin(__builtin_amdgcn_rcpf)
  return __builtin_amdgcn_rcpf(1.f + __expf(-x));   // v_exp_f32 + v_rcp_f32
#else
  return 1.f / (1.f + __expf(-x));
#endif
}
__device__ inline float fast_tanh(float x) {
#if defined(__has_builtin) && __has_builtin(__builtin_amdgcn_tanhf)
  return __builtin_amdgcn_tanhf(x);                 // v_tanh_f32
#else
  return 2.f * fast_sigmoid(2.f * x) - 1.f;         // branch-free fallback
#endif
}

__device__ inline unsigned pack_bf16(float a, float b) {
  union { unsigned u; __bf16 h[2]; } p;
  p.h[0] = (__bf16)a;
  p.h[1] = (__bf16)b;
  return p.u;
}

#if CDNA5_HAS_TDM
// One 2-D TDM descriptor: tile of `rows` rows x 64 bytes from a tensor whose
// dim0 row is `stride` bytes, into contiguous LDS at ldsAddr.
__device__ inline void tdm_load_2d(unsigned ldsAddr, unsigned long long gaddr,
                                   unsigned rows, unsigned stride) {
  v4u g0;
  g0[0] = 1u;                                            // count=1 valid user D#
  g0[1] = ldsAddr;                                       // LDS byte address
  g0[2] = (unsigned)gaddr;                               // global addr [31:0]
  g0[3] = (unsigned)((gaddr >> 32) & 0x01FFFFFFu) | (2u << 30);  // [56:32] | type=2
  v8i g1;
  g1[0] = 0;                                             // no multicast, 1B elems
  g1[1] = (int)((stride & 0xFFFFu) << 16);               // tensor_dim0[15:0]
  g1[2] = (int)((stride >> 16) | (rows << 16));          // td0[31:16] | td1[15:0]
  g1[3] = (int)(64u << 16);                              // td1[31:16]=0 | tile_dim0=64B
  g1[4] = (int)rows;                                     // tile_dim1=rows, tile_dim2=0
  g1[5] = (int)stride;                                   // dim0_stride[31:0]
  g1[6] = 0;                                             // dim0_stride[47:32]
  g1[7] = 0;
  v4i z4 = {};
#if __clang_major__ >= 23
  v8i z8 = {};
  __builtin_amdgcn_tensor_load_to_lds(g0, g1, z4, z4, z8, 0);
#else
  __builtin_amdgcn_tensor_load_to_lds(g0, g1, z4, z4, 0);
#endif
}
#endif

// ---------------------------------------------------------------------------
// Weight repack to bf16 GEMM operands (kappa pairs land on dword boundaries).
// ---------------------------------------------------------------------------
__global__ void k_prep(const float* __restrict__ fw, const float* __restrict__ gw,
                       const float* __restrict__ rw, const float* __restrict__ sw,
                       bf16* __restrict__ Wfg, bf16* __restrict__ Wr, bf16* __restrict__ Ws) {
  int idx = blockIdx.x * blockDim.x + threadIdx.x;

  const int nfg = NLAYER * 64 * 96;      // [i][m][tap*32+rc]
  if (idx < nfg) {
    int kk = idx % 96, m = (idx / 96) % 64, i = idx / (96 * 64);
    int tap = kk >> 5, rc = kk & 31;
    float v = (m < 32) ? fw[((i * 32 + m) * 32 + rc) * 3 + tap]
                       : gw[((i * 32 + (m - 32)) * 32 + rc) * 3 + tap];
    Wfg[idx] = (bf16)v;
  }
  const int nr = NLAYER * 32 * 32;       // [i][m][k]
  if (idx < nr) Wr[idx] = (bf16)rw[idx];

  const int ns = SC * NLAYER * 32;       // [sc][i*32+dc]
  if (idx < ns) {
    int k = idx % (NLAYER * 32), sc = idx / (NLAYER * 32);
    int i = k >> 5, dc = k & 31;
    Ws[idx] = (bf16)sw[(i * SC + sc) * DC + dc];
  }
}

// 1x1 start conv -> fp32 residual stream X and pair-interleaved bf16 copy Xb
__global__ void k_start(const float* __restrict__ in, const float* __restrict__ sw,
                        float* __restrict__ X, bf16* __restrict__ Xb) {
  int idx = blockIdx.x * blockDim.x + threadIdx.x;
  if (idx >= RC * NTOT) return;
  int n = idx & (NTOT - 1), rc = idx >> 16;
  int b = n & 63, l = n >> 6;
  float v = in[b * 2 * LEN + l] * sw[rc * 2 + 0] +
            in[b * 2 * LEN + LEN + l] * sw[rc * 2 + 1];
  X[idx] = v;
  Xb[((size_t)(rc >> 1) * NTOT + n) * 2 + (rc & 1)] = (bf16)v;
}

// ---------------------------------------------------------------------------
// One WaveNet layer. 4 waves/block, each wave owns a 16-column tile.
//  gated:  acc[64][16] = sum over 3 taps of Wfg[64][0:32] x Xg_tap[32][16]  (branch-free clamp)
//  h = tanh(f)*sigmoid(g) -> Hcat (pair-interleaved bf16) + LDS
//  res:    Xout = Wr[32][32] x h + Xin  (Xin fed as the WMMA C operand)
// ---------------------------------------------------------------------------
__global__ __launch_bounds__(128)
void k_layer(const float* __restrict__ Xin, const bf16* __restrict__ Xbin,
             float* __restrict__ Xout, bf16* __restrict__ Xbout,
             bf16* __restrict__ Hl, const bf16* __restrict__ Wfg_i,
             const bf16* __restrict__ Wr_i, int dil) {
  __shared__ unsigned hbuf[4][16][16];   // packed h pairs [wave][c2][col]
  const int lane = threadIdx.x & 31;
  const int wv   = threadIdx.x >> 5;
  const int wid  = blockIdx.x * 4 + wv;
  const int m16  = lane & 15;
  const int half = lane >> 4;
  const int col  = wid * 16 + m16;

  const unsigned* XbU = (const unsigned*)Xbin;   // [c2][n] dwords
  unsigned*       HbU = (unsigned*)Hl;
  unsigned*       XoU = (unsigned*)Xbout;

  // A fragments: 4 m-tiles (64 rows = filter|gate), K=32 (cols 0..31 of 96, reused for all taps); dword pair loads
  FragBF afrag[4];
#pragma unroll
  for (int mt = 0; mt < 4; ++mt) {
#pragma unroll
    for (int v = 0; v < 8; ++v) {
      int k0 = half * 16 + v * 2;
      afrag[mt].u[v] = *(const unsigned*)(Wfg_i + (mt * 16 + m16) * 96 + k0);
    }
  }

  v8f acc[4];
#pragma unroll
  for (int mt = 0; mt < 4; ++mt)
#pragma unroll
    for (int j = 0; j < 8; ++j) acc[mt][j] = 0.f;

#pragma unroll
  for (int tap = 0; tap < 3; ++tap) {
    int  shift = (tap - 1) * dil * 64;
    int  sn    = col + shift;
    bool ok    = (unsigned)sn < (unsigned)NTOT;
    int  snc   = sn < 0 ? 0 : (sn > NTOT - 1 ? NTOT - 1 : sn);  // clamped addr
    FragBF bfrag;
#pragma unroll
    for (int v = 0; v < 8; ++v) {
      unsigned u = XbU[(size_t)(half * 8 + v) * NTOT + snc];    // unconditional
      bfrag.u[v] = ok ? u : 0u;                                 // cndmask, no branch
    }
#pragma unroll
    for (int mt = 0; mt < 4; ++mt)
      acc[mt] = WMMA_BF16(afrag[mt].bf, bfrag.bf, acc[mt]);
  }

  // gated nonlinearity; C/D layout: vgpr v -> m = half*8 + v. Pack row pairs.
#pragma unroll
  for (int v = 0; v < 8; v += 2) {
    int m0 = half * 8 + v;           // rows m0, m0+1 -> c2 = m0>>1
    float h0a = fast_tanh(acc[0][v])     * fast_sigmoid(acc[2][v]);
    float h0b = fast_tanh(acc[0][v + 1]) * fast_sigmoid(acc[2][v + 1]);
    float h1a = fast_tanh(acc[1][v])     * fast_sigmoid(acc[3][v]);
    float h1b = fast_tanh(acc[1][v + 1]) * fast_sigmoid(acc[3][v + 1]);
    unsigned u0 = pack_bf16(h0a, h0b);   // channels m0,  m0+1
    unsigned u1 = pack_bf16(h1a, h1b);   // channels m0+16, m0+17
    int c2 = m0 >> 1;
    HbU[(size_t)c2 * NTOT + col]       = u0;
    HbU[(size_t)(8 + c2) * NTOT + col] = u1;
    hbuf[wv][c2][m16]     = u0;
    hbuf[wv][8 + c2][m16] = u1;
  }
  __syncthreads();

  // residual GEMM: B fragment of h straight from packed LDS
  FragBF bh;
#pragma unroll
  for (int v = 0; v < 8; ++v) bh.u[v] = hbuf[wv][half * 8 + v][m16];

#pragma unroll
  for (int mt = 0; mt < 2; ++mt) {
    FragBF ar;
#pragma unroll
    for (int v = 0; v < 8; ++v) {
      int k0 = half * 16 + v * 2;
      ar.u[v] = *(const unsigned*)(Wr_i + (mt * 16 + m16) * 32 + k0);
    }
    v8f c;
#pragma unroll
    for (int v = 0; v < 8; ++v)
      c[v] = Xin[(size_t)(mt * 16 + half * 8 + v) * NTOT + col];  // residual as C
    c = WMMA_BF16(ar.bf, bh.bf, c);
#pragma unroll
    for (int v = 0; v < 8; v += 2) {
      int m0 = mt * 16 + half * 8 + v;
      Xout[(size_t)m0 * NTOT + col]       = c[v];
      Xout[(size_t)(m0 + 1) * NTOT + col] = c[v + 1];
      XoU[(size_t)(m0 >> 1) * NTOT + col] = pack_bf16(c[v], c[v + 1]);
    }
  }
}

// ---------------------------------------------------------------------------
// skip GEMM: rskip[256][N] = relu( Wskip_cat[256][1280] x Hcat[1280][N] )
// One wave / 16-column tile, 16 live accumulator tiles (128 acc VGPRs).
// The 640x64B Hcat panel is staged per-wave into LDS by the Tensor Data Mover
// (one tensor_load_to_lds + s_wait_tensorcnt), then consumed as ds_load_b32.
// ---------------------------------------------------------------------------
__global__ __launch_bounds__(128)
void k_skip(const bf16* __restrict__ H, const bf16* __restrict__ Ws,
            bf16* __restrict__ rskip) {
  extern __shared__ unsigned lds_h[];    // 4 waves x 640 rows x 16 dwords
  const int lane = threadIdx.x & 31;
  const int wv   = threadIdx.x >> 5;
  const int wid  = blockIdx.x * 4 + wv;
  const int m16  = lane & 15;
  const int half = lane >> 4;
  const int col  = wid * 16 + m16;

#if CDNA5_HAS_TDM
  const unsigned wbase = (unsigned)wv * 640u * 16u;
  {
    unsigned ldsAddr = (unsigned)(uintptr_t)&lds_h[wbase];
    unsigned long long ga =
        (unsigned long long)(uintptr_t)H + (unsigned long long)(wid * 16) * 4ull;
    tdm_load_2d(ldsAddr, ga, 640u, (unsigned)(NTOT * 4));  // rows of 64B
    __builtin_amdgcn_s_wait_tensorcnt(0);
  }
#else
  const unsigned* HU = (const unsigned*)H;
#endif

  v8f acc[16];
#pragma unroll
  for (int mt = 0; mt < 16; ++mt)
#pragma unroll
    for (int j = 0; j < 8; ++j) acc[mt][j] = 0.f;

  for (int ks = 0; ks < NLAYER; ++ks) {
    FragBF bfrag;
#pragma unroll
    for (int v = 0; v < 8; ++v) {
      int c2 = ks * 16 + half * 8 + v;
#if CDNA5_HAS_TDM
      bfrag.u[v] = lds_h[wbase + (unsigned)c2 * 16u + (unsigned)m16];
#else
      bfrag.u[v] = HU[(size_t)c2 * NTOT + col];
#endif
    }
#pragma unroll
    for (int mt = 0; mt < 16; ++mt) {
      FragBF a;
#pragma unroll
      for (int v = 0; v < 8; ++v) {
        int k0 = ks * 32 + half * 16 + v * 2;
        a.u[v] = *(const unsigned*)(Ws + (size_t)(mt * 16 + m16) * (NLAYER * 32) + k0);
      }
      acc[mt] = WMMA_BF16(a.bf, bfrag.bf, acc[mt]);
    }
  }
#pragma unroll
  for (int mt = 0; mt < 16; ++mt)
#pragma unroll
    for (int v = 0; v < 8; ++v) {
      float r = acc[mt][v];
      r = r > 0.f ? r : 0.f;  // fused ReLU
      rskip[(size_t)(mt * 16 + half * 8 + v) * NTOT + col] = (bf16)r;
    }
}

// ---------------------------------------------------------------------------
// FC: out[64][256] = relu_skip[64][262144] x fc_w^T, split-K (64 chunks).
// block = 4 waves: 64 batch rows x 16 classes, K chunk 4096. fc_w staged
// fp32->bf16 pair-packed through LDS with coalesced float4 loads + prefetch.
// ---------------------------------------------------------------------------
__global__ __launch_bounds__(128)
void k_fc(const bf16* __restrict__ rskip, const float* __restrict__ fcw,
          float* __restrict__ partial) {
  __shared__ __align__(16) bf16 lA[32][64];      // [k][b]
  __shared__ unsigned lBp[16][16];               // packed k-pairs x class
  const int t    = threadIdx.x;
  const int lane = t & 31, wv = t >> 5;
  const int half = lane >> 4, n16 = lane & 15;
  const int ct   = blockIdx.x & 15;   // class tile (16 classes)
  const int s    = blockIdx.x >> 4;   // K split (64 splits)
  const int clsbase = ct * 16;
  const size_t kbase = (size_t)s * 4096;

  v8f acc;
#pragma unroll
  for (int j = 0; j < 8; ++j) acc[j] = 0.f;

  const int clsld = t >> 3;          // 0..15: class row this thread stages
  const int kq    = (t & 7) * 4;     // 0..28: 4 consecutive k's

  for (int ks = 0; ks < 128; ++ks) {
    const size_t k0 = kbase + (size_t)ks * 32;
    const int sc = (int)(k0 >> 10);
    const int l0 = (int)(k0 & 1023);

    // A tile: rskip[sc][l0*64 .. +2048) is contiguous == lA[k][b]
    const uint4* srcA = (const uint4*)(rskip + (size_t)sc * NTOT + (size_t)l0 * 64);
    uint4* dA = (uint4*)&lA[0][0];
    dA[t]       = srcA[t];
    dA[t + 128] = srcA[t + 128];

    // B tile: fc_w rows [clsbase..+16), cols [k0..k0+32), fp32 -> packed bf16
    const float4 wf = *(const float4*)(fcw + (size_t)(clsbase + clsld) * KFC + k0 + kq);
    lBp[(kq >> 1) + 0][clsld] = pack_bf16(wf.x, wf.y);
    lBp[(kq >> 1) + 1][clsld] = pack_bf16(wf.z, wf.w);
    if (ks + 1 < 128)
      __builtin_prefetch(fcw + (size_t)(clsbase + clsld) * KFC + k0 + 32 + kq, 0, 1);
    __syncthreads();

    FragBF a, b;
#pragma unroll
    for (int v = 0; v < 8; ++v) {
      int ka = half * 16 + v * 2;
      a.bf[2 * v]     = lA[ka][wv * 16 + n16];
      a.bf[2 * v + 1] = lA[ka + 1][wv * 16 + n16];
      b.u[v]          = lBp[half * 8 + v][n16];
    }
    acc = WMMA_BF16(a.bf, b.bf, acc);
    __syncthreads();
  }

#pragma unroll
  for (int v = 0; v < 8; ++v) {
    int bb = wv * 16 + half * 8 + v;
    partial[((size_t)s * 64 + bb) * CLS + clsbase + n16] = acc[v];
  }
}

// deterministic split-K reduction + bias
__global__ void k_reduce(const float* __restrict__ partial, const float* __restrict__ fcb,
                         float* __restrict__ out) {
  int idx = blockIdx.x * blockDim.x + threadIdx.x;
  if (idx >= BATCH * CLS) return;
  int b = idx >> 8, c = idx & 255;
  float sum = fcb[c];
  for (int sp = 0; sp < 64; ++sp) sum += partial[((size_t)sp * 64 + b) * CLS + c];
  out[idx] = sum;
}

// ---------------------------------------------------------------------------
extern "C" void kernel_launch(void* const* d_in, const int* in_sizes, int n_in,
                              void* d_out, int out_size, void* d_ws, size_t ws_size,
                              hipStream_t stream) {
  (void)in_sizes; (void)n_in; (void)out_size; (void)ws_size;
  const float* input    = (const float*)d_in[0];
  const float* start_w  = (const float*)d_in[1];
  const float* filter_w = (const float*)d_in[2];
  const float* gate_w   = (const float*)d_in[3];
  const float* res_w    = (const float*)d_in[4];
  const float* skip_w   = (const float*)d_in[5];
  const float* fc_w     = (const float*)d_in[6];
  const float* fc_b     = (const float*)d_in[7];

  // Workspace carve (~230 MB): X fp32 ping-pong 16MB, Xb bf16 ping-pong 8MB,
  // Hcat 160MB, rskip 32MB, bf16 weights ~1.2MB, split-K partials 4MB.
  char* ws = (char*)d_ws;
  auto carve = [&](size_t bytes) -> char* {
    char* p = ws;
    ws += (bytes + 255) & ~(size_t)255;
    return p;
  };
  float* X0   = (float*)carve(sizeof(float) * (size_t)RC * NTOT);
  float* X1   = (float*)carve(sizeof(float) * (size_t)RC * NTOT);
  bf16*  Xb0  = (bf16*)carve(sizeof(bf16) * (size_t)RC * NTOT);
  bf16*  Xb1  = (bf16*)carve(sizeof(bf16) * (size_t)RC * NTOT);
  bf16*  Hcat = (bf16*)carve(sizeof(bf16) * (size_t)NLAYER * 32 * NTOT);
  bf16*  Wfg  = (bf16*)carve(sizeof(bf16) * (size_t)NLAYER * 64 * 96);
  bf16*  Wr   = (bf16*)carve(sizeof(bf16) * (size_t)NLAYER * 32 * 32);
  bf16*  Wsk  = (bf16*)carve(sizeof(bf16) * (size_t)SC * NLAYER * 32);
  bf16*  rsk  = (bf16*)carve(sizeof(bf16) * (size_t)SC * NTOT);
  float* part = (float*)carve(sizeof(float) * (size_t)64 * BATCH * CLS);

  k_prep<<<(SC * NLAYER * 32 + 255) / 256, 256, 0, stream>>>(
      filter_w, gate_w, res_w, skip_w, Wfg, Wr, Wsk);
  k_start<<<(RC * NTOT) / 256, 256, 0, stream>>>(input, start_w, X0, Xb0);

  float* xin = X0;  float* xout = X1;
  bf16*  bin = Xb0; bf16*  bout = Xb1;
  for (int i = 0; i < NLAYER; ++i) {
    int dil = 1 << (i % 10);  // dilation schedule: 1..512 per block of 10 layers
    k_layer<<<NTOT / 64, 128, 0, stream>>>(
        xin, bin, xout, bout, Hcat + (size_t)i * 32 * NTOT,
        Wfg + (size_t)i * 64 * 96, Wr + (size_t)i * 32 * 32, dil);
    float* tf = xin; xin = xout; xout = tf;
    bf16*  tb = bin; bin = bout; bout = tb;
  }

  k_skip<<<NTOT / 64, 128, 4 * 640 * 16 * sizeof(unsigned), stream>>>(Hcat, Wsk, rsk);
  k_fc<<<1024, 128, 0, stream>>>(rsk, fc_w, part);
  k_reduce<<<(BATCH * CLS) / 256, 256, 0, stream>>>(part, fc_b, (float*)d_out);
}
